// NeighborEmbedding_25855703122101
// MI455X (gfx1250) — compile-verified, matrix operand order checked by reading, not run on hardware
//
#include <hip/hip_runtime.h>
#include <hip/hip_bf16.h>

typedef __attribute__((ext_vector_type(16))) _Float16 v16h;
typedef __attribute__((ext_vector_type(8)))  float    v8f;
typedef __attribute__((ext_vector_type(4)))  int      i32x4_;

#define B_  16
#define N_  4096
#define S1_ 512
#define S2_ 256
#define K_  32
#define EPS_ 1e-5f

#if __has_builtin(__builtin_amdgcn_global_load_async_to_lds_b128)
#define HAVE_ASYNC_LDS 1
#else
#define HAVE_ASYNC_LDS 0
#endif

// ---------------------------------------------------------------------------
// xyz[b,n,c] = x[b,c,n]
__global__ void k_xyz_from_x(const float* __restrict__ x, float* __restrict__ xyz) {
  int id = blockIdx.x * blockDim.x + threadIdx.x;           // over B*N
  if (id >= B_ * N_) return;
  int b = id / N_, n = id % N_;
#pragma unroll
  for (int c = 0; c < 3; ++c)
    xyz[(size_t)id * 3 + c] = x[((size_t)b * 3 + c) * N_ + n];
}

// y1[p,o] = sum_d w1[o,d] * x[b,d,n]   (p = b*N+n), raw pre-BN fp32
__global__ void k_cbr1(const float* __restrict__ x, const float* __restrict__ w1,
                       float* __restrict__ y1) {
  int p = blockIdx.x * blockDim.x + threadIdx.x;
  if (p >= B_ * N_) return;
  int b = p / N_, n = p % N_;
  float x0 = x[((size_t)b * 3 + 0) * N_ + n];
  float x1 = x[((size_t)b * 3 + 1) * N_ + n];
  float x2 = x[((size_t)b * 3 + 2) * N_ + n];
  float* yr = y1 + (size_t)p * 64;
#pragma unroll 4
  for (int o = 0; o < 64; ++o)
    yr[o] = w1[o * 3 + 0] * x0 + w1[o * 3 + 1] * x1 + w1[o * 3 + 2] * x2;
}

// ---------------------------------------------------------------------------
// Column statistics (population mean / invstd) over (Mrows, C) matrices.
__global__ void k_colstats_f32(const float* __restrict__ Y, int Mrows, int C,
                               float* __restrict__ stats) {
  int c = blockIdx.x, t = threadIdx.x;
  float s = 0.f, s2 = 0.f;
  for (int i = t; i < Mrows; i += 256) {
    float v = Y[(size_t)i * C + c];
    s += v; s2 += v * v;
  }
  __shared__ float sh[256], sh2[256];
  sh[t] = s; sh2[t] = s2; __syncthreads();
  for (int o = 128; o > 0; o >>= 1) {
    if (t < o) { sh[t] += sh[t + o]; sh2[t] += sh2[t + o]; }
    __syncthreads();
  }
  if (t == 0) {
    float m = sh[0] / (float)Mrows;
    float var = sh2[0] / (float)Mrows - m * m;
    stats[c] = m;
    stats[C + c] = rsqrtf(fmaxf(var, 0.f) + EPS_);
  }
}

__global__ void k_colstats_f16(const _Float16* __restrict__ Y, int Mrows, int C,
                               float* __restrict__ stats) {
  int c = blockIdx.x, t = threadIdx.x;
  float s = 0.f, s2 = 0.f;
  for (int i = t; i < Mrows; i += 256) {
    float v = (float)Y[(size_t)i * C + c];
    s += v; s2 += v * v;
  }
  __shared__ float sh[256], sh2[256];
  sh[t] = s; sh2[t] = s2; __syncthreads();
  for (int o = 128; o > 0; o >>= 1) {
    if (t < o) { sh[t] += sh[t + o]; sh2[t] += sh2[t + o]; }
    __syncthreads();
  }
  if (t == 0) {
    float m = sh[0] / (float)Mrows;
    float var = sh2[0] / (float)Mrows - m * m;
    stats[c] = m;
    stats[C + c] = rsqrtf(fmaxf(var, 0.f) + EPS_);
  }
}

__global__ void k_bnrelu_f32_f16(const float* __restrict__ Y, const float* __restrict__ stats,
                                 const float* __restrict__ g, const float* __restrict__ bt,
                                 long long total, int C, _Float16* __restrict__ outp) {
  long long id = (long long)blockIdx.x * blockDim.x + threadIdx.x;
  if (id >= total) return;
  int c = (int)(id % C);
  float v = (Y[id] - stats[c]) * stats[C + c] * g[c] + bt[c];
  outp[id] = (_Float16)fmaxf(v, 0.f);
}

__global__ void k_bnrelu_f16_ip(_Float16* __restrict__ Y, const float* __restrict__ stats,
                                const float* __restrict__ g, const float* __restrict__ bt,
                                long long total, int C) {
  long long id = (long long)blockIdx.x * blockDim.x + threadIdx.x;
  if (id >= total) return;
  int c = (int)(id % C);
  float v = ((float)Y[id] - stats[c]) * stats[C + c] * g[c] + bt[c];
  Y[id] = (_Float16)fmaxf(v, 0.f);
}

__global__ void k_f32_to_f16(const float* __restrict__ in, _Float16* __restrict__ outp, int n) {
  int id = blockIdx.x * blockDim.x + threadIdx.x;
  if (id < n) outp[id] = (_Float16)in[id];
}

// ---------------------------------------------------------------------------
// WMMA GEMM: Y[M,O] (f16) = A[M,Kd] (f16, row-major) x W[O,Kd]^T (f16, row-major)
//
// Block = 256 threads = 8 waves. Block tile = 128(M) x 64(O).
// The 64-column W strip (64 x Kd f16, <=32KB) is staged into LDS once per
// block (global_load_async_to_lds_b128 when available, tracked by ASYNCcnt),
// then each wave owns a 16x64 output tile: one A fragment is reused across
// 4 WMMAs whose B fragments come from LDS (ds_load_b128 pairs).
//
// gfx1250 16-bit A/B fragment layout: lane<16 holds K = {k0..k0+7, k0+16..k0+23},
// lane>=16 holds K = {k0+8..k0+15, k0+24..k0+31}; row/col = lane & 15.
__global__ void k_gemm_wmma(const _Float16* __restrict__ A,
                            const _Float16* __restrict__ W,
                            _Float16* __restrict__ Y,
                            int M, int Kd, int O) {
  __shared__ _Float16 Wst[64 * 256];          // 32 KB max (Kd <= 256)

  const int lane = threadIdx.x & 31;
  const int wave = threadIdx.x >> 5;
  const int oBlocks = O >> 6;                 // O / 64
  const int bM = blockIdx.x / oBlocks;
  const int bO = blockIdx.x - bM * oBlocks;

  // ---- stage W strip [bO*64, bO*64+64) x Kd into LDS ----
  {
    const int chunks = (64 * Kd) >> 3;        // 16-byte chunks
    const _Float16* wbase = W + (size_t)bO * 64 * Kd;
#if HAVE_ASYNC_LDS
    for (int i = threadIdx.x; i < chunks; i += 256) {
      __builtin_amdgcn_global_load_async_to_lds_b128(
          (i32x4_ __attribute__((address_space(1)))*)(wbase + (size_t)i * 8),
          (i32x4_ __attribute__((address_space(3)))*)(&Wst[i * 8]),
          0, 0);
    }
#if __has_builtin(__builtin_amdgcn_s_wait_asynccnt)
    __builtin_amdgcn_s_wait_asynccnt(0);
#else
    asm volatile("s_wait_asynccnt 0x0" ::: "memory");
#endif
#else
    for (int i = threadIdx.x; i < chunks; i += 256)
      ((uint4*)Wst)[i] = ((const uint4*)wbase)[i];
#endif
    __syncthreads();
  }

  const int r  = lane & 15;
  const int kb = (lane >> 4) << 3;            // 0 or 8

  const _Float16* arow = A + (size_t)(bM * 128 + wave * 16 + r) * Kd + kb;

  v8f acc[4];
#pragma unroll
  for (int j = 0; j < 4; ++j) acc[j] = (v8f){};

  for (int k0 = 0; k0 < Kd; k0 += 32) {
    v16h af;
#pragma unroll
    for (int j = 0; j < 8; ++j) {
      af[j]     = arow[k0 + j];
      af[j + 8] = arow[k0 + 16 + j];
    }
    __builtin_prefetch(arow + k0 + 64, 0, 1);
#pragma unroll
    for (int j = 0; j < 4; ++j) {
      v16h bf;
      const _Float16* wrow = &Wst[(size_t)(j * 16 + r) * Kd + k0 + kb];
#pragma unroll
      for (int e = 0; e < 8; ++e) {
        bf[e]     = wrow[e];
        bf[e + 8] = wrow[16 + e];
      }
      acc[j] = __builtin_amdgcn_wmma_f32_16x16x32_f16(false, af, false, bf,
                                                      (short)0, acc[j], false, false);
    }
  }

  const int col   = lane & 15;
  const int rbase = (lane < 16) ? 0 : 8;
  const int rowb  = bM * 128 + wave * 16 + rbase;
#pragma unroll
  for (int j = 0; j < 4; ++j) {
    _Float16* yout = Y + (size_t)rowb * O + (size_t)bO * 64 + j * 16 + col;
#pragma unroll
    for (int rr = 0; rr < 8; ++rr)
      yout[(size_t)rr * O] = (_Float16)acc[j][rr];
  }
}

// ---------------------------------------------------------------------------
// Farthest-point sampling: one workgroup (256 threads) per batch, dist in LDS.
__global__ void k_fps(const float* __restrict__ coords, int Np, int S,
                      int* __restrict__ outIdx) {
  __shared__ float dist[4096];
  __shared__ float rval[256];
  __shared__ int   ridx[256];
  __shared__ int   s_far;
  const int b = blockIdx.x, t = threadIdx.x;
  const float* cb = coords + (size_t)b * Np * 3;
  for (int i = t; i < Np; i += 256) dist[i] = 1e10f;
  if (t == 0) s_far = 0;
  __syncthreads();
  for (int s = 0; s < S; ++s) {
    int far = s_far;
    if (t == 0) outIdx[b * S + s] = far;
    float cx = cb[far * 3], cy = cb[far * 3 + 1], cz = cb[far * 3 + 2];
    float bv = -1.f; int bi = 0x7fffffff;
    for (int i = t; i < Np; i += 256) {
      float dx = cb[i * 3] - cx, dy = cb[i * 3 + 1] - cy, dz = cb[i * 3 + 2] - cz;
      float d  = dx * dx + dy * dy + dz * dz;
      float nd = fminf(dist[i], d);
      dist[i] = nd;
      if (nd > bv || (nd == bv && i < bi)) { bv = nd; bi = i; }
    }
    rval[t] = bv; ridx[t] = bi;
    __syncthreads();
    for (int o = 128; o > 0; o >>= 1) {
      if (t < o) {
        if (rval[t + o] > rval[t] || (rval[t + o] == rval[t] && ridx[t + o] < ridx[t])) {
          rval[t] = rval[t + o]; ridx[t] = ridx[t + o];
        }
      }
      __syncthreads();
    }
    if (t == 0) s_far = ridx[0];
    __syncthreads();
  }
}

__global__ void k_gather_xyz(const float* __restrict__ xyz, const int* __restrict__ idx,
                             int S, int Np, float* __restrict__ outp) {
  int id = blockIdx.x * blockDim.x + threadIdx.x;            // over B*S
  if (id >= B_ * S) return;
  int b = id / S;
  int src = idx[id];
#pragma unroll
  for (int c = 0; c < 3; ++c)
    outp[(size_t)id * 3 + c] = xyz[((size_t)b * Np + src) * 3 + c];
}

__global__ void k_gather_feat(const _Float16* __restrict__ feat, const int* __restrict__ idx,
                              int S, int Np, int C, _Float16* __restrict__ outp) {
  long long id = (long long)blockIdx.x * blockDim.x + threadIdx.x;  // over B*S*C
  if (id >= (long long)B_ * S * C) return;
  int c  = (int)(id % C);
  int bs = (int)(id / C);
  int b  = bs / S;
  int src = idx[bs];
  outp[id] = feat[((size_t)b * Np + src) * C + c];
}

// ---------------------------------------------------------------------------
// kNN (K_=32 smallest sq-distances): one workgroup per (b,s) row,
// distances in LDS, 32 rounds of block-argmin.
__global__ void k_knn(const float* __restrict__ q, const float* __restrict__ p,
                      int S, int Np, int* __restrict__ knn) {
  __shared__ float dist[4096];
  __shared__ float rval[256];
  __shared__ int   ridx[256];
  const int row = blockIdx.x;           // b*S + s
  const int b = row / S, t = threadIdx.x;
  const float* qp = q + (size_t)row * 3;
  const float qx = qp[0], qy = qp[1], qz = qp[2];
  const float* pb = p + (size_t)b * Np * 3;
  for (int i = t; i < Np; i += 256) {
    float dx = pb[i * 3] - qx, dy = pb[i * 3 + 1] - qy, dz = pb[i * 3 + 2] - qz;
    dist[i] = dx * dx + dy * dy + dz * dz;
  }
  __syncthreads();
  for (int k = 0; k < K_; ++k) {
    float bv = 3.4e38f; int bi = 0x7fffffff;
    for (int i = t; i < Np; i += 256) {
      float d = dist[i];
      if (d < bv || (d == bv && i < bi)) { bv = d; bi = i; }
    }
    rval[t] = bv; ridx[t] = bi;
    __syncthreads();
    for (int o = 128; o > 0; o >>= 1) {
      if (t < o) {
        if (rval[t + o] < rval[t] || (rval[t + o] == rval[t] && ridx[t + o] < ridx[t])) {
          rval[t] = rval[t + o]; ridx[t] = ridx[t + o];
        }
      }
      __syncthreads();
    }
    if (t == 0) {
      knn[(size_t)row * K_ + k] = ridx[0];
      dist[ridx[0]] = 3.4e38f;
    }
    __syncthreads();
  }
}

// agg[m, 0:Cin]    = feat[b, knn, c] - center[bs, c]
// agg[m, Cin:2Cin] = center[bs, c]          (m = (b*S+s)*K_ + k)
__global__ void k_agg(const _Float16* __restrict__ feat, const _Float16* __restrict__ cfeat,
                      const int* __restrict__ knn, int S, int Np, int Cin,
                      _Float16* __restrict__ agg, long long total) {
  long long id = (long long)blockIdx.x * blockDim.x + threadIdx.x;   // over M*Cin
  if (id >= total) return;
  int       c  = (int)(id % Cin);
  long long m  = id / Cin;
  int       bs = (int)(m / K_);
  int       k  = (int)(m % K_);
  int       b  = bs / S;
  int nIdx = knn[(size_t)bs * K_ + k];
  float gv = (float)feat[((size_t)b * Np + nIdx) * Cin + c];
  float cv = (float)cfeat[(size_t)bs * Cin + c];
  _Float16* arow = agg + (size_t)m * 2 * Cin;
  arow[c]       = (_Float16)(gv - cv);
  arow[Cin + c] = (_Float16)cv;
}

// max over K_ axis: (BS, K_, C) f16 -> (BS, C) f16
__global__ void k_maxk_f16(const _Float16* __restrict__ Y, int BS, int C,
                           _Float16* __restrict__ outp) {
  long long id = (long long)blockIdx.x * blockDim.x + threadIdx.x;   // over BS*C
  if (id >= (long long)BS * C) return;
  int c = (int)(id % C);
  long long bs = id / C;
  float m = -3.4e38f;
  for (int k = 0; k < K_; ++k)
    m = fmaxf(m, (float)Y[((size_t)bs * K_ + k) * C + c]);
  outp[id] = (_Float16)m;
}

// final: (B,S,K_,C) f16 -> out (B,C,S) fp32
__global__ void k_maxk_out_t(const _Float16* __restrict__ Y, int S, int C,
                             float* __restrict__ outp) {
  long long id = (long long)blockIdx.x * blockDim.x + threadIdx.x;   // over B*S*C
  if (id >= (long long)B_ * S * C) return;
  int c = (int)(id % C);
  int s = (int)((id / C) % S);
  int b = (int)(id / ((long long)S * C));
  long long bs = (long long)b * S + s;
  float m = -3.4e38f;
  for (int k = 0; k < K_; ++k)
    m = fmaxf(m, (float)Y[((size_t)bs * K_ + k) * C + c]);
  outp[((size_t)b * C + c) * S + s] = m;
}

// ---------------------------------------------------------------------------
extern "C" void kernel_launch(void* const* d_in, const int* in_sizes, int n_in,
                              void* d_out, int out_size, void* d_ws, size_t ws_size,
                              hipStream_t stream) {
  const float* x    = (const float*)d_in[0];
  const float* w1   = (const float*)d_in[1];
  const float* w2   = (const float*)d_in[2];
  const float* g1   = (const float*)d_in[3];
  const float* b1   = (const float*)d_in[4];
  const float* g2   = (const float*)d_in[5];
  const float* b2   = (const float*)d_in[6];
  const float* s1w1 = (const float*)d_in[7];
  const float* s1w2 = (const float*)d_in[8];
  const float* s1g1 = (const float*)d_in[9];
  const float* s1b1 = (const float*)d_in[10];
  const float* s1g2 = (const float*)d_in[11];
  const float* s1b2 = (const float*)d_in[12];
  const float* s2w1 = (const float*)d_in[13];
  const float* s2w2 = (const float*)d_in[14];
  const float* s2g1 = (const float*)d_in[15];
  const float* s2b1 = (const float*)d_in[16];
  const float* s2g2 = (const float*)d_in[17];
  const float* s2b2 = (const float*)d_in[18];
  float* out = (float*)d_out;

  char* ws = (char*)d_ws;
  size_t off = 0;
  auto alloc = [&](size_t bytes) -> void* {
    void* p = ws + off;
    off = (off + bytes + 255) & ~(size_t)255;
    return p;
  };
  auto cdiv = [](long long a, long long b) -> int { return (int)((a + b - 1) / b); };

  const long long M1 = (long long)B_ * S1_ * K_;   // 262144
  const long long M2 = (long long)B_ * S2_ * K_;   // 131072

  float*     xyz    = (float*)alloc((size_t)B_ * N_ * 3 * 4);
  float*     y1     = (float*)alloc((size_t)B_ * N_ * 64 * 4);
  _Float16*  h0     = (_Float16*)alloc((size_t)B_ * N_ * 64 * 2);
  _Float16*  fbuf   = (_Float16*)alloc((size_t)B_ * N_ * 64 * 2);
  float*     stats  = (float*)alloc(512 * 4);
  _Float16*  w2h    = (_Float16*)alloc(64 * 64 * 2);
  _Float16*  s1w1h  = (_Float16*)alloc(128 * 128 * 2);
  _Float16*  s1w2h  = (_Float16*)alloc(128 * 128 * 2);
  _Float16*  s2w1h  = (_Float16*)alloc(256 * 256 * 2);
  _Float16*  s2w2h  = (_Float16*)alloc(256 * 256 * 2);
  int*       fidx1  = (int*)alloc((size_t)B_ * S1_ * 4);
  float*     nxyz1  = (float*)alloc((size_t)B_ * S1_ * 3 * 4);
  _Float16*  nfeat1 = (_Float16*)alloc((size_t)B_ * S1_ * 64 * 2);
  int*       knn1   = (int*)alloc((size_t)B_ * S1_ * K_ * 4);
  _Float16*  agg    = (_Float16*)alloc((size_t)M1 * 128 * 2);   // == M2*256*2
  _Float16*  hid    = (_Float16*)alloc((size_t)M1 * 128 * 2);
  _Float16*  f1     = (_Float16*)alloc((size_t)B_ * S1_ * 128 * 2);
  int*       fidx2  = (int*)alloc((size_t)B_ * S2_ * 4);
  float*     nxyz2  = (float*)alloc((size_t)B_ * S2_ * 3 * 4);
  _Float16*  nfeat2 = (_Float16*)alloc((size_t)B_ * S2_ * 128 * 2);
  int*       knn2   = (int*)alloc((size_t)B_ * S2_ * K_ * 4);
  (void)ws_size; (void)in_sizes; (void)n_in; (void)out_size;

  // ---- weight conversions ----
  k_f32_to_f16<<<cdiv(64 * 64, 256), 256, 0, stream>>>(w2, w2h, 64 * 64);
  k_f32_to_f16<<<cdiv(128 * 128, 256), 256, 0, stream>>>(s1w1, s1w1h, 128 * 128);
  k_f32_to_f16<<<cdiv(128 * 128, 256), 256, 0, stream>>>(s1w2, s1w2h, 128 * 128);
  k_f32_to_f16<<<cdiv(256 * 256, 256), 256, 0, stream>>>(s2w1, s2w1h, 256 * 256);
  k_f32_to_f16<<<cdiv(256 * 256, 256), 256, 0, stream>>>(s2w2, s2w2h, 256 * 256);

  // ---- cbr stage ----
  k_xyz_from_x<<<cdiv((long long)B_ * N_, 256), 256, 0, stream>>>(x, xyz);
  k_cbr1<<<cdiv((long long)B_ * N_, 256), 256, 0, stream>>>(x, w1, y1);
  k_colstats_f32<<<64, 256, 0, stream>>>(y1, B_ * N_, 64, stats);
  k_bnrelu_f32_f16<<<cdiv((long long)B_ * N_ * 64, 256), 256, 0, stream>>>(
      y1, stats, g1, b1, (long long)B_ * N_ * 64, 64, h0);
  k_gemm_wmma<<<(B_ * N_ / 128) * (64 / 64), 256, 0, stream>>>(h0, w2h, fbuf, B_ * N_, 64, 64);
  k_colstats_f16<<<64, 256, 0, stream>>>(fbuf, B_ * N_, 64, stats);
  k_bnrelu_f16_ip<<<cdiv((long long)B_ * N_ * 64, 256), 256, 0, stream>>>(
      fbuf, stats, g2, b2, (long long)B_ * N_ * 64, 64);

  // ---- sg block 1 ----
  k_fps<<<B_, 256, 0, stream>>>(xyz, N_, S1_, fidx1);
  k_gather_xyz<<<cdiv((long long)B_ * S1_, 256), 256, 0, stream>>>(xyz, fidx1, S1_, N_, nxyz1);
  k_gather_feat<<<cdiv((long long)B_ * S1_ * 64, 256), 256, 0, stream>>>(
      fbuf, fidx1, S1_, N_, 64, nfeat1);
  k_knn<<<B_ * S1_, 256, 0, stream>>>(nxyz1, xyz, S1_, N_, knn1);
  k_agg<<<cdiv(M1 * 64, 256), 256, 0, stream>>>(fbuf, nfeat1, knn1, S1_, N_, 64, agg, M1 * 64);
  {
    int grid = (int)(M1 / 128) * (128 / 64);
    k_gemm_wmma<<<grid, 256, 0, stream>>>(agg, s1w1h, hid, (int)M1, 128, 128);
    k_colstats_f16<<<128, 256, 0, stream>>>(hid, (int)M1, 128, stats);
    k_bnrelu_f16_ip<<<cdiv(M1 * 128, 256), 256, 0, stream>>>(hid, stats, s1g1, s1b1, M1 * 128, 128);
    k_gemm_wmma<<<grid, 256, 0, stream>>>(hid, s1w2h, agg, (int)M1, 128, 128);
    k_colstats_f16<<<128, 256, 0, stream>>>(agg, (int)M1, 128, stats);
    k_bnrelu_f16_ip<<<cdiv(M1 * 128, 256), 256, 0, stream>>>(agg, stats, s1g2, s1b2, M1 * 128, 128);
  }
  k_maxk_f16<<<cdiv((long long)B_ * S1_ * 128, 256), 256, 0, stream>>>(agg, B_ * S1_, 128, f1);

  // ---- sg block 2 ----
  k_fps<<<B_, 256, 0, stream>>>(nxyz1, S1_, S2_, fidx2);
  k_gather_xyz<<<cdiv((long long)B_ * S2_, 256), 256, 0, stream>>>(nxyz1, fidx2, S2_, S1_, nxyz2);
  k_gather_feat<<<cdiv((long long)B_ * S2_ * 128, 256), 256, 0, stream>>>(
      f1, fidx2, S2_, S1_, 128, nfeat2);
  k_knn<<<B_ * S2_, 256, 0, stream>>>(nxyz2, nxyz1, S2_, S1_, knn2);
  k_agg<<<cdiv(M2 * 128, 256), 256, 0, stream>>>(f1, nfeat2, knn2, S2_, S1_, 128, agg, M2 * 128);
  {
    int grid = (int)(M2 / 128) * (256 / 64);
    k_gemm_wmma<<<grid, 256, 0, stream>>>(agg, s2w1h, hid, (int)M2, 256, 256);
    k_colstats_f16<<<256, 256, 0, stream>>>(hid, (int)M2, 256, stats);
    k_bnrelu_f16_ip<<<cdiv(M2 * 256, 256), 256, 0, stream>>>(hid, stats, s2g1, s2b1, M2 * 256, 256);
    k_gemm_wmma<<<grid, 256, 0, stream>>>(hid, s2w2h, agg, (int)M2, 256, 256);
    k_colstats_f16<<<256, 256, 0, stream>>>(agg, (int)M2, 256, stats);
    k_bnrelu_f16_ip<<<cdiv(M2 * 256, 256), 256, 0, stream>>>(agg, stats, s2g2, s2b2, M2 * 256, 256);
  }
  k_maxk_out_t<<<cdiv((long long)B_ * S2_ * 256, 256), 256, 0, stream>>>(agg, S2_, 256, out);
}
